// AttentionBlock_36661840839814
// MI455X (gfx1250) — matrix operand order repaired and checked by reading, not compiled, not run on hardware
//
#include <hip/hip_runtime.h>
#include <hip/hip_fp16.h>

typedef _Float16 h16;
typedef __attribute__((ext_vector_type(2)))  _Float16 h2;
typedef __attribute__((ext_vector_type(8)))  _Float16 v8h;
typedef __attribute__((ext_vector_type(16))) _Float16 v16h;
typedef __attribute__((ext_vector_type(8)))  float    v8f;

#define DIM   512
#define QKVD  1536
#define SEQ   4096
#define NTOK  8192      // B*N
#define HEADS 8
#define HD    64

// LDS row strides (halves) padded for bank spread
#define KLD_STRIDE 72   // 32x64 K tile rows
#define VLD_STRIDE 40   // 64x32 transposed V tile rows
#define PLD_STRIDE 36   // 16x32 P tile rows

#define WMMA_F16(A, B, C) \
  __builtin_amdgcn_wmma_f32_16x16x32_f16(false, (A), false, (B), (short)0, (C), false, false)

// ---- fragment loaders (ISA 7.12.2 layouts, wave32) -------------------------
// A 16x32 f16: lane l -> g=l>>4, m=l&15. VGPR p (halves 2p,2p+1):
//   p<4 : K = g*8 + 2p ; p>=4 : K = 16 + g*8 + 2(p-4)
__device__ __forceinline__ v16h load_a_frag(const h16* rowp, int g) {
  v16h a;
#pragma unroll
  for (int p = 0; p < 8; ++p) {
    int K = (p < 4) ? (g * 8 + 2 * p) : (16 + g * 8 + 2 * (p - 4));
    h2 v = *(const h2*)(rowp + K);
    a[2 * p]     = v[0];
    a[2 * p + 1] = v[1];
  }
  return a;
}
// B 32x16 f16, K contiguous at colp: lane -> n=l&15; VGPR p holds K (same as A),+1
__device__ __forceinline__ v16h load_b_frag_kmajor(const h16* colp, int g) {
  v16h b;
#pragma unroll
  for (int p = 0; p < 8; ++p) {
    int K = (p < 4) ? (g * 8 + 2 * p) : (16 + g * 8 + 2 * (p - 4));
    h2 v = *(const h2*)(colp + K);
    b[2 * p]     = v[0];
    b[2 * p + 1] = v[1];
  }
  return b;
}

// ---- 1) LayerNorm: one wave per row of 512 ---------------------------------
__global__ void ln_kernel(const float* __restrict__ x, const float* __restrict__ gamma,
                          const float* __restrict__ beta, h16* __restrict__ xn) {
  int row  = blockIdx.x * 8 + (threadIdx.x >> 5);
  int lane = threadIdx.x & 31;
  const float* xr = x + (size_t)row * DIM;
  float vals[16];
  float s = 0.f, ss = 0.f;
#pragma unroll
  for (int e = 0; e < 16; ++e) {
    float v = xr[e * 32 + lane];
    vals[e] = v; s += v; ss += v * v;
  }
#pragma unroll
  for (int off = 16; off >= 1; off >>= 1) {
    s  += __shfl_xor(s, off);
    ss += __shfl_xor(ss, off);
  }
  float mu  = s * (1.f / DIM);
  float var = ss * (1.f / DIM) - mu * mu;
  float inv = rsqrtf(var + 1e-5f);
  h16* o = xn + (size_t)row * DIM;
#pragma unroll
  for (int e = 0; e < 16; ++e) {
    int c = e * 32 + lane;
    o[c] = (h16)((vals[e] - mu) * inv * gamma[c] + beta[c]);
  }
}

// ---- 2) convert weights to f16, transposed (K-major per output column) ----
__global__ void cvt_kernel(const float* __restrict__ wqkv, const float* __restrict__ wproj,
                           h16* __restrict__ wqkvT, h16* __restrict__ wprojT) {
  int idx0   = blockIdx.x * blockDim.x + threadIdx.x;
  int stride = gridDim.x * blockDim.x;
  for (int i = idx0; i < DIM * QKVD; i += stride) {
    int k = i / QKVD, n = i % QKVD;
    wqkvT[(size_t)n * DIM + k] = (h16)wqkv[i];
  }
  for (int i = idx0; i < DIM * DIM; i += stride) {
    int k = i / DIM, n = i % DIM;
    wprojT[(size_t)n * DIM + k] = (h16)wproj[i];
  }
}

// ---- 3) QKV GEMM: 16x64 tile per wave, A-frag reused 4x --------------------
__global__ void gemm_qkv_kernel(const h16* __restrict__ A, const h16* __restrict__ BT,
                                const float* __restrict__ bias, h16* __restrict__ C) {
  int wid  = blockIdx.x * 8 + (threadIdx.x >> 5);
  int lane = threadIdx.x & 31;
  int g = lane >> 4, m = lane & 15;
  const int NT4 = QKVD / 64;              // 24 supertiles of 64 cols
  int tm = wid / NT4, tn = wid % NT4;
  const h16* arow = A + (size_t)(tm * 16 + m) * DIM;
  const h16* bcol[4];
#pragma unroll
  for (int j = 0; j < 4; ++j)
    bcol[j] = BT + (size_t)(tn * 64 + j * 16 + m) * DIM;
  v8f acc0{}, acc1{}, acc2{}, acc3{};
  for (int k0 = 0; k0 < DIM; k0 += 32) {
    __builtin_prefetch(arow + k0 + 64, 0, 0);
    v16h a  = load_a_frag(arow + k0, g);
    v16h b0 = load_b_frag_kmajor(bcol[0] + k0, g);
    v16h b1 = load_b_frag_kmajor(bcol[1] + k0, g);
    v16h b2 = load_b_frag_kmajor(bcol[2] + k0, g);
    v16h b3 = load_b_frag_kmajor(bcol[3] + k0, g);
    acc0 = WMMA_F16(a, b0, acc0);
    acc1 = WMMA_F16(a, b1, acc1);
    acc2 = WMMA_F16(a, b2, acc2);
    acc3 = WMMA_F16(a, b3, acc3);
  }
  h16* outr = C + (size_t)(tm * 16 + 8 * g) * QKVD + tn * 64 + m;
#pragma unroll
  for (int j = 0; j < 4; ++j) {
    float bv = bias[tn * 64 + j * 16 + m];
    v8f* accp = (j == 0) ? &acc0 : (j == 1) ? &acc1 : (j == 2) ? &acc2 : &acc3;
#pragma unroll
    for (int r = 0; r < 8; ++r)
      outr[(size_t)r * QKVD + j * 16] = (h16)((*accp)[r] + bv);
  }
}

// ---- 4) flash attention: 8 waves/block share (b,h); K/V staged in LDS ------
__global__ void attn_kernel(const h16* __restrict__ qkv, h16* __restrict__ out) {
  __shared__ __align__(16) h16 klds[32 * KLD_STRIDE];        // K block, row=key
  __shared__ __align__(16) h16 vldsT[64 * VLD_STRIDE];       // V block, row=d
  __shared__ __align__(16) h16 plds[8][16 * PLD_STRIDE];     // per-wave P tile
  int tid  = threadIdx.x;
  int wav  = tid >> 5;
  int lane = tid & 31;
  int g = lane >> 4, m = lane & 15;
  int bid = blockIdx.x;                 // 512 blocks: 2 * 8 * 32
  int bb = bid >> 8;                    // batch
  int hh = (bid >> 5) & 7;              // head
  int qg = bid & 31;                    // query group (8 tiles)
  int qt = qg * 8 + wav;                // this wave's 16-query tile
  const h16* base  = qkv + (size_t)bb * SEQ * QKVD;
  const h16* qrow  = base + (size_t)(qt * 16 + m) * QKVD + hh * HD;
  const h16* kbase = base + 512  + hh * HD;
  const h16* vbase = base + 1024 + hh * HD;
  v16h qa0 = load_a_frag(qrow, g);        // d = 0..31
  v16h qa1 = load_a_frag(qrow + 32, g);   // d = 32..63
  v8f o0{}, o1{}, o2{}, o3{};
  float mrun[8], lrun[8];
#pragma unroll
  for (int r = 0; r < 8; ++r) { mrun[r] = -1e30f; lrun[r] = 0.f; }

  int crow = tid >> 3;   // 0..31: cooperative key row
  int cch  = tid & 7;    // 0..7 : 8-half chunk along d

  for (int kb = 0; kb < SEQ; kb += 32) {
    // cooperative global->LDS staging of K (row-major) and V (transposed)
    v8h kv = *(const v8h*)(kbase + (size_t)(kb + crow) * QKVD + cch * 8);
    v8h vv = *(const v8h*)(vbase + (size_t)(kb + crow) * QKVD + cch * 8);
    __syncthreads();                          // prior iteration's reads done
    *(v8h*)(klds + crow * KLD_STRIDE + cch * 8) = kv;
#pragma unroll
    for (int i = 0; i < 8; ++i)
      vldsT[(cch * 8 + i) * VLD_STRIDE + crow] = vv[i];
    __syncthreads();                          // tiles visible to all waves

    // S = Q K^T (two 16-key tiles), B frags from LDS (K rows are K-major)
    const h16* kr0 = klds + (size_t)m * KLD_STRIDE;
    const h16* kr1 = klds + (size_t)(16 + m) * KLD_STRIDE;
    v16h b00 = load_b_frag_kmajor(kr0, g);
    v16h b01 = load_b_frag_kmajor(kr0 + 32, g);
    v16h b10 = load_b_frag_kmajor(kr1, g);
    v16h b11 = load_b_frag_kmajor(kr1 + 32, g);
    v8f s0{}, s1{};
    s0 = WMMA_F16(qa0, b00, s0);
    s0 = WMMA_F16(qa1, b01, s0);
    s1 = WMMA_F16(qa0, b10, s1);
    s1 = WMMA_F16(qa1, b11, s1);

    // online softmax; C-layout row r lives on 16-lane half g, cols = lanes
#pragma unroll
    for (int r = 0; r < 8; ++r) {
      float a0 = s0[r] * 0.125f;     // 1/sqrt(64)
      float a1 = s1[r] * 0.125f;
      float t = fmaxf(a0, a1);
      t = fmaxf(t, __shfl_xor(t, 1));
      t = fmaxf(t, __shfl_xor(t, 2));
      t = fmaxf(t, __shfl_xor(t, 4));
      t = fmaxf(t, __shfl_xor(t, 8));
      float nm = fmaxf(mrun[r], t);
      float f  = __expf(mrun[r] - nm);
      mrun[r]  = nm;
      float p0 = __expf(a0 - nm);
      float p1 = __expf(a1 - nm);
      float rs = p0 + p1;
      rs += __shfl_xor(rs, 1);
      rs += __shfl_xor(rs, 2);
      rs += __shfl_xor(rs, 4);
      rs += __shfl_xor(rs, 8);
      lrun[r] = lrun[r] * f + rs;
      o0[r] = o0[r] * f; o1[r] = o1[r] * f;
      o2[r] = o2[r] * f; o3[r] = o3[r] * f;
      int rr = r + 8 * g;
      plds[wav][rr * PLD_STRIDE + m]      = (h16)p0;   // keys kb..kb+15
      plds[wav][rr * PLD_STRIDE + 16 + m] = (h16)p1;   // keys kb+16..kb+31
    }

    // reload P as A-fragment (C-layout -> A-layout shuffle via LDS, same wave)
    v16h pa;
#pragma unroll
    for (int p = 0; p < 8; ++p) {
      int K = (p < 4) ? (g * 8 + 2 * p) : (16 + g * 8 + 2 * (p - 4));
      pa[2 * p]     = plds[wav][m * PLD_STRIDE + K];
      pa[2 * p + 1] = plds[wav][m * PLD_STRIDE + K + 1];
    }

    // O += P V : B frags from transposed V in LDS (K=key contiguous per d row)
    v16h vb0 = load_b_frag_kmajor(vldsT + (size_t)(0  + m) * VLD_STRIDE, g);
    v16h vb1 = load_b_frag_kmajor(vldsT + (size_t)(16 + m) * VLD_STRIDE, g);
    v16h vb2 = load_b_frag_kmajor(vldsT + (size_t)(32 + m) * VLD_STRIDE, g);
    v16h vb3 = load_b_frag_kmajor(vldsT + (size_t)(48 + m) * VLD_STRIDE, g);
    o0 = WMMA_F16(pa, vb0, o0);
    o1 = WMMA_F16(pa, vb1, o1);
    o2 = WMMA_F16(pa, vb2, o2);
    o3 = WMMA_F16(pa, vb3, o3);
  }

  // epilogue: normalize and store f16 in (token, h*64+d) layout
  h16* orow = out + (size_t)(bb * SEQ + qt * 16 + 8 * g) * DIM + hh * HD + m;
#pragma unroll
  for (int r = 0; r < 8; ++r) {
    float inv = 1.f / lrun[r];
    orow[(size_t)r * DIM + 0]  = (h16)(o0[r] * inv);
    orow[(size_t)r * DIM + 16] = (h16)(o1[r] * inv);
    orow[(size_t)r * DIM + 32] = (h16)(o2[r] * inv);
    orow[(size_t)r * DIM + 48] = (h16)(o3[r] * inv);
  }
}

// ---- 5) out projection: 16x64 tile per wave -> f32 d_out + bias ------------
__global__ void gemm_proj_kernel(const h16* __restrict__ A, const h16* __restrict__ BT,
                                 const float* __restrict__ bias, float* __restrict__ C) {
  int wid  = blockIdx.x * 8 + (threadIdx.x >> 5);
  int lane = threadIdx.x & 31;
  int g = lane >> 4, m = lane & 15;
  const int NT4 = DIM / 64;               // 8 supertiles of 64 cols
  int tm = wid / NT4, tn = wid % NT4;
  const h16* arow = A + (size_t)(tm * 16 + m) * DIM;
  const h16* bcol[4];
#pragma unroll
  for (int j = 0; j < 4; ++j)
    bcol[j] = BT + (size_t)(tn * 64 + j * 16 + m) * DIM;
  v8f acc0{}, acc1{}, acc2{}, acc3{};
  for (int k0 = 0; k0 < DIM; k0 += 32) {
    __builtin_prefetch(arow + k0 + 64, 0, 0);
    v16h a  = load_a_frag(arow + k0, g);
    v16h b0 = load_b_frag_kmajor(bcol[0] + k0, g);
    v16h b1 = load_b_frag_kmajor(bcol[1] + k0, g);
    v16h b2 = load_b_frag_kmajor(bcol[2] + k0, g);
    v16h b3 = load_b_frag_kmajor(bcol[3] + k0, g);
    acc0 = WMMA_F16(a, b0, acc0);
    acc1 = WMMA_F16(a, b1, acc1);
    acc2 = WMMA_F16(a, b2, acc2);
    acc3 = WMMA_F16(a, b3, acc3);
  }
  float* outr = C + (size_t)(tm * 16 + 8 * g) * DIM + tn * 64 + m;
#pragma unroll
  for (int j = 0; j < 4; ++j) {
    float bv = bias[tn * 64 + j * 16 + m];
    v8f* accp = (j == 0) ? &acc0 : (j == 1) ? &acc1 : (j == 2) ? &acc2 : &acc3;
#pragma unroll
    for (int r = 0; r < 8; ++r)
      outr[(size_t)r * DIM + j * 16] = (*accp)[r] + bv;
  }
}

// ---------------------------------------------------------------------------
extern "C" void kernel_launch(void* const* d_in, const int* in_sizes, int n_in,
                              void* d_out, int out_size, void* d_ws, size_t ws_size,
                              hipStream_t stream) {
  const float* x     = (const float*)d_in[0];
  const float* gamma = (const float*)d_in[1];
  const float* beta  = (const float*)d_in[2];
  const float* wqkv  = (const float*)d_in[3];
  const float* bqkv  = (const float*)d_in[4];
  const float* wproj = (const float*)d_in[5];
  const float* bproj = (const float*)d_in[6];
  float* out = (float*)d_out;

  char* ws = (char*)d_ws;
  h16* xn     = (h16*)(ws);                            //  8 MiB: 8192x512 f16
  h16* wqkvT  = (h16*)(ws + (size_t) 8 * 1024 * 1024); // 1.5 MiB: 1536x512 f16
  h16* wprojT = (h16*)(ws + (size_t)10 * 1024 * 1024); // 0.5 MiB: 512x512 f16
  h16* qkvbuf = (h16*)(ws + (size_t)11 * 1024 * 1024); // 24 MiB: 8192x1536 f16
  h16* attnbf = (h16*)(ws + (size_t)36 * 1024 * 1024); //  8 MiB: 8192x512 f16

  ln_kernel<<<NTOK / 8, 256, 0, stream>>>(x, gamma, beta, xn);
  cvt_kernel<<<768, 256, 0, stream>>>(wqkv, wproj, wqkvT, wprojT);
  // 512 row-tiles * 24 col-supertiles, 8 waves/block
  gemm_qkv_kernel<<<(NTOK / 16) * (QKVD / 64) / 8, 256, 0, stream>>>(xn, wqkvT, bqkv, qkvbuf);
  // 2 batches * 8 heads * 32 query-groups
  attn_kernel<<<2 * HEADS * (SEQ / 128), 256, 0, stream>>>(qkvbuf, attnbf);
  // 512 row-tiles * 8 col-supertiles, 8 waves/block
  gemm_proj_kernel<<<(NTOK / 16) * (DIM / 64) / 8, 256, 0, stream>>>(attnbf, wprojT, bproj, out);
}
